// AttentativePoolingLayer_7687991460478
// MI455X (gfx1250) — matrix operand order repaired and checked by reading, not compiled, not run on hardware
//
#include <hip/hip_runtime.h>
#include <math.h>

// ---------------------------------------------------------------------------
// AttentativePooling for MI455X (gfx1250, wave32, WMMA)
//   align = tanh(A^T U B) + mask ; softmax(max rows/cols) ; weighted pools
// bf16 WMMA 16x16x32 (f32 accum) for both GEMMs. 128x128 macro-tile, 8 waves,
// 8 tiles/wave. LDS slabs laid out so every fragment is two ds_load_b128.
// Transposed slab staging uses 4x4 register micro-transpose -> ds_store_b64.
// tanh+mask+row/col-max fused into GEMM-2 epilogue (align never materialized).
// ---------------------------------------------------------------------------

typedef __bf16 bf16_t;
typedef __attribute__((ext_vector_type(16))) __bf16    v16bf;
typedef __attribute__((ext_vector_type(8)))  float     v8f;
typedef __attribute__((ext_vector_type(4)))  float     f32x4;
typedef __attribute__((ext_vector_type(4)))  unsigned  u32x4;
typedef __attribute__((ext_vector_type(2)))  unsigned  u32x2;

#define BSZ   16
#define DIM   768
#define SEQA  1024
#define SEQB  1024
#define KSTEP 32
#define TM    128           // macro-tile M
#define TN    128           // macro-tile N
#define LPAD  40            // LDS row stride in bf16 (80B = 20 banks, conflict-free)

union FragU  { v16bf v; u32x4 q[2]; unsigned u[8]; bf16_t h[16]; };
union Pack4  { bf16_t h[4]; u32x2 u2; };

// Monotone order-preserving encoding of float into unsigned (for atomic max).
__device__ __forceinline__ unsigned encf(float x) {
  unsigned b = __float_as_uint(x);
  return (b & 0x80000000u) ? ~b : (b | 0x80000000u);
}
__device__ __forceinline__ float decf(unsigned e) {
  unsigned b = (e & 0x80000000u) ? (e ^ 0x80000000u) : ~e;
  return __uint_as_float(b);
}

__global__ void init_kernel(unsigned* __restrict__ p, int n) {
  int i = blockIdx.x * blockDim.x + threadIdx.x;
  if (i < n) p[i] = 0u;  // strictly below encf(-inf)
}

// ---------------------------------------------------------------------------
// GEMM 1: Cbf[b,m,n] = bf16( sum_k U[m,k] * B[b,k,n] )   M=768 N=1024 K=768
// ---------------------------------------------------------------------------
__global__ void __launch_bounds__(256)
gemm_ub_kernel(const float* __restrict__ U, const float* __restrict__ Bm,
               bf16_t* __restrict__ Cbf) {
  __shared__ bf16_t Asl[TM][LPAD];    // [m][k]  (K contiguous per row)
  __shared__ bf16_t Bsl[TN][LPAD];    // [n][k]  (K contiguous per row)
  const int b  = blockIdx.z;
  const int m0 = blockIdx.x * TM;
  const int n0 = blockIdx.y * TN;
  const int tid  = threadIdx.x;
  const int lane = tid & 31, wid = tid >> 5;
  const int wm = wid >> 2, wn = wid & 3;     // 2(M) x 4(N) wave grid
  const int half = lane >> 4, l15 = lane & 15;
  const float* Bb = Bm + (size_t)b * DIM * SEQB;
  const int kg4 = (tid >> 5) * 4, ng4 = (tid & 31) * 4;  // 4x4 transpose block

  v8f acc[4][2];
#pragma unroll
  for (int i = 0; i < 4; ++i)
#pragma unroll
    for (int j = 0; j < 2; ++j) acc[i][j] = v8f{};

  for (int k0 = 0; k0 < DIM; k0 += KSTEP) {
    __syncthreads();
    // Stage U slab: 128 rows x 32 k, f32x4 loads, packed b64 LDS stores.
#pragma unroll
    for (int it = 0; it < (TM * KSTEP) / (256 * 4); ++it) {
      int q = tid + it * 256;
      int m = q >> 3, k = (q & 7) * 4;
      f32x4 src = *(const f32x4*)&U[(size_t)(m0 + m) * DIM + (k0 + k)];
      Pack4 p;
#pragma unroll
      for (int j = 0; j < 4; ++j) p.h[j] = (bf16_t)src[j];
      *(u32x2*)&Asl[m][k] = p.u2;
    }
    // Stage B slab transposed into [n][k]: 4x4 register micro-transpose,
    // 4 coalesced b128 global loads -> 4 packed ds_store_b64.
    {
      Pack4 col[4];
#pragma unroll
      for (int kk = 0; kk < 4; ++kk) {
        f32x4 src = *(const f32x4*)&Bb[(size_t)(k0 + kg4 + kk) * SEQB + (n0 + ng4)];
#pragma unroll
        for (int j = 0; j < 4; ++j) col[j].h[kk] = (bf16_t)src[j];
      }
#pragma unroll
      for (int j = 0; j < 4; ++j) *(u32x2*)&Bsl[ng4 + j][kg4] = col[j].u2;
    }
    if (k0 + KSTEP < DIM)  // global_prefetch_b8 hint for next K-slab
      __builtin_prefetch(&Bb[(size_t)(k0 + KSTEP + (tid >> 3)) * SEQB + n0], 0, 1);
    __syncthreads();

    // Fragments: each is two ds_load_b128 (k 8*half..+7 and 16+8*half..+7 per lane).
    FragU bfr[2];
#pragma unroll
    for (int j = 0; j < 2; ++j) {
      const int nrow = wn * 32 + j * 16 + l15;
      bfr[j].q[0] = *(const u32x4*)&Bsl[nrow][half * 8];
      bfr[j].q[1] = *(const u32x4*)&Bsl[nrow][16 + half * 8];
    }
#pragma unroll
    for (int i = 0; i < 4; ++i) {
      const int mrow = wm * 64 + i * 16 + l15;
      FragU af;
      af.q[0] = *(const u32x4*)&Asl[mrow][half * 8];
      af.q[1] = *(const u32x4*)&Asl[mrow][16 + half * 8];
#pragma unroll
      for (int j = 0; j < 2; ++j)
        acc[i][j] = __builtin_amdgcn_wmma_f32_16x16x32_bf16(
            false, af.v, false, bfr[j].v, (short)0, acc[i][j], false, false);
    }
  }

#pragma unroll
  for (int i = 0; i < 4; ++i) {
    int mt = wm * 64 + i * 16;
#pragma unroll
    for (int j = 0; j < 2; ++j) {
      int nt = wn * 32 + j * 16;
#pragma unroll
      for (int v = 0; v < 8; ++v) {
        int m = m0 + mt + v + 8 * half;
        int n = n0 + nt + l15;
        Cbf[((size_t)b * DIM + m) * SEQB + n] = (bf16_t)acc[i][j][v];
      }
    }
  }
}

// ---------------------------------------------------------------------------
// GEMM 2 (fused): align[s,t] = tanh( sum_d A[b,d,s]*C[b,d,t] ) + mask[b,s,t]
// Epilogue reduces row-max (over t) and col-max (over s) via encoded atomics.
// ---------------------------------------------------------------------------
__global__ void __launch_bounds__(256)
gemm_align_kernel(const float* __restrict__ Ain, const bf16_t* __restrict__ Cbf,
                  const float* __restrict__ Msk,
                  unsigned* __restrict__ rowMaxEnc,
                  unsigned* __restrict__ colMaxEnc) {
  __shared__ bf16_t Asl[TM][LPAD];    // [s][d]
  __shared__ bf16_t Bsl[TN][LPAD];    // [t][d]
  const int b  = blockIdx.z;
  const int s0 = blockIdx.x * TM;
  const int t0 = blockIdx.y * TN;
  const int tid  = threadIdx.x;
  const int lane = tid & 31, wid = tid >> 5;
  const int wm = wid >> 2, wn = wid & 3;
  const int half = lane >> 4, l15 = lane & 15;
  const float*  Ab = Ain + (size_t)b * DIM * SEQA;
  const bf16_t* Cb = Cbf + (size_t)b * DIM * SEQB;
  const int kg4 = (tid >> 5) * 4, ng4 = (tid & 31) * 4;

  v8f acc[4][2];
#pragma unroll
  for (int i = 0; i < 4; ++i)
#pragma unroll
    for (int j = 0; j < 2; ++j) acc[i][j] = v8f{};

  for (int k0 = 0; k0 < DIM; k0 += KSTEP) {
    __syncthreads();
    // A^T slab [s][d]: 4x4 micro-transpose (global rows are d, contiguous s).
    {
      Pack4 col[4];
#pragma unroll
      for (int kk = 0; kk < 4; ++kk) {
        f32x4 src = *(const f32x4*)&Ab[(size_t)(k0 + kg4 + kk) * SEQA + (s0 + ng4)];
#pragma unroll
        for (int j = 0; j < 4; ++j) col[j].h[kk] = (bf16_t)src[j];
      }
#pragma unroll
      for (int j = 0; j < 4; ++j) *(u32x2*)&Asl[ng4 + j][kg4] = col[j].u2;
    }
    // C slab [t][d]: bf16 source, 4x4 micro-transpose, b64 in / b64 out.
    {
      Pack4 col[4];
#pragma unroll
      for (int kk = 0; kk < 4; ++kk) {
        Pack4 row;
        row.u2 = *(const u32x2*)&Cb[(size_t)(k0 + kg4 + kk) * SEQB + (t0 + ng4)];
#pragma unroll
        for (int j = 0; j < 4; ++j) col[j].h[kk] = row.h[j];
      }
#pragma unroll
      for (int j = 0; j < 4; ++j) *(u32x2*)&Bsl[ng4 + j][kg4] = col[j].u2;
    }
    if (k0 + KSTEP < DIM)
      __builtin_prefetch(&Cb[(size_t)(k0 + KSTEP + (tid >> 3)) * SEQB + t0], 0, 1);
    __syncthreads();

    FragU bfr[2];
#pragma unroll
    for (int j = 0; j < 2; ++j) {
      const int nrow = wn * 32 + j * 16 + l15;
      bfr[j].q[0] = *(const u32x4*)&Bsl[nrow][half * 8];
      bfr[j].q[1] = *(const u32x4*)&Bsl[nrow][16 + half * 8];
    }
#pragma unroll
    for (int i = 0; i < 4; ++i) {
      const int mrow = wm * 64 + i * 16 + l15;
      FragU af;
      af.q[0] = *(const u32x4*)&Asl[mrow][half * 8];
      af.q[1] = *(const u32x4*)&Asl[mrow][16 + half * 8];
#pragma unroll
      for (int j = 0; j < 2; ++j)
        acc[i][j] = __builtin_amdgcn_wmma_f32_16x16x32_bf16(
            false, af.v, false, bfr[j].v, (short)0, acc[i][j], false, false);
    }
  }

  // Fused epilogue: tanh + mask, then row/col max reductions + atomics.
#pragma unroll
  for (int i = 0; i < 4; ++i) {
    int mt = wm * 64 + i * 16;
#pragma unroll
    for (int j = 0; j < 2; ++j) {
      int nt = wn * 32 + j * 16;
      const int tg = t0 + nt + l15;
      float vals[8];
#pragma unroll
      for (int v = 0; v < 8; ++v) {
        int sg = s0 + mt + v + 8 * half;
        vals[v] = tanhf(acc[i][j][v]) +
                  Msk[((size_t)b * SEQA + sg) * SEQB + tg];
      }
      // Row max: reduce over the 16 lanes of this half-wave (t direction).
#pragma unroll
      for (int v = 0; v < 8; ++v) {
        float r = vals[v];
        r = fmaxf(r, __shfl_xor(r, 1));
        r = fmaxf(r, __shfl_xor(r, 2));
        r = fmaxf(r, __shfl_xor(r, 4));
        r = fmaxf(r, __shfl_xor(r, 8));
        if (l15 == 0) {
          int sg = s0 + mt + v + 8 * half;
          atomicMax(&rowMaxEnc[b * SEQA + sg], encf(r));
        }
      }
      // Col max: local max over 8 rows, combine the two halves (s direction).
      float c = vals[0];
#pragma unroll
      for (int v = 1; v < 8; ++v) c = fmaxf(c, vals[v]);
      c = fmaxf(c, __shfl_xor(c, 16));
      if (half == 0) atomicMax(&colMaxEnc[b * SEQB + tg], encf(c));
    }
  }
}

// ---------------------------------------------------------------------------
// Softmax over 1024 encoded maxima per (batch, side).
// ---------------------------------------------------------------------------
__global__ void __launch_bounds__(256)
softmax_kernel(const unsigned* __restrict__ rowMaxEnc,
               const unsigned* __restrict__ colMaxEnc,
               float* __restrict__ scoreA, float* __restrict__ scoreB) {
  const int b = blockIdx.x, which = blockIdx.y;
  const unsigned* src = which ? colMaxEnc : rowMaxEnc;
  float* dst = which ? scoreB : scoreA;
  const int n = 1024;
  __shared__ float sm[8];
  const int tid = threadIdx.x, lane = tid & 31, wid = tid >> 5;

  float vals[4], m = -3.4e38f;
#pragma unroll
  for (int i = 0; i < 4; ++i) {
    vals[i] = decf(src[b * n + tid + i * 256]);
    m = fmaxf(m, vals[i]);
  }
#pragma unroll
  for (int o = 1; o < 32; o <<= 1) m = fmaxf(m, __shfl_xor(m, o));
  if (lane == 0) sm[wid] = m;
  __syncthreads();
  float bm = sm[0];
#pragma unroll
  for (int w = 1; w < 8; ++w) bm = fmaxf(bm, sm[w]);

  float e[4], s = 0.f;
#pragma unroll
  for (int i = 0; i < 4; ++i) { e[i] = expf(vals[i] - bm); s += e[i]; }
#pragma unroll
  for (int o = 1; o < 32; o <<= 1) s += __shfl_xor(s, o);
  __syncthreads();
  if (lane == 0) sm[wid] = s;
  __syncthreads();
  float bs = 0.f;
#pragma unroll
  for (int w = 0; w < 8; ++w) bs += sm[w];
  float inv = 1.0f / bs;
#pragma unroll
  for (int i = 0; i < 4; ++i) dst[b * n + tid + i * 256] = e[i] * inv;
}

// ---------------------------------------------------------------------------
// out[z,b,d] = sum_s in[b,d,s] * score[b,s]
// ---------------------------------------------------------------------------
__global__ void __launch_bounds__(256)
pool_kernel(const float* __restrict__ Ain, const float* __restrict__ Bin,
            const float* __restrict__ scoreA, const float* __restrict__ scoreB,
            float* __restrict__ out) {
  const int d = blockIdx.x, b = blockIdx.y, z = blockIdx.z;
  const float* in = z ? Bin : Ain;
  const float* sc = z ? scoreB : scoreA;
  __shared__ float sm[8];
  const int tid = threadIdx.x, lane = tid & 31, wid = tid >> 5;
  const float* row = in + ((size_t)b * DIM + d) * 1024;
  const float* sv  = sc + (size_t)b * 1024;
  float p = 0.f;
  for (int i = tid; i < 1024; i += 256) p += row[i] * sv[i];
#pragma unroll
  for (int o = 16; o > 0; o >>= 1) p += __shfl_xor(p, o);
  if (lane == 0) sm[wid] = p;
  __syncthreads();
  if (tid == 0) {
    float t = 0.f;
#pragma unroll
    for (int w = 0; w < 8; ++w) t += sm[w];
    out[(size_t)z * BSZ * DIM + (size_t)b * DIM + d] = t;
  }
}

// ---------------------------------------------------------------------------
extern "C" void kernel_launch(void* const* d_in, const int* in_sizes, int n_in,
                              void* d_out, int out_size, void* d_ws, size_t ws_size,
                              hipStream_t stream) {
  (void)in_sizes; (void)n_in; (void)out_size; (void)ws_size;
  const float* Ain = (const float*)d_in[0];   // (16, 768, 1024)
  const float* Bin = (const float*)d_in[1];   // (16, 768, 1024)
  const float* Msk = (const float*)d_in[2];   // (16, 1024, 1024)
  const float* U   = (const float*)d_in[3];   // (768, 768)
  float* out = (float*)d_out;                 // [output_A | output_B] flat

  char* ws = (char*)d_ws;
  bf16_t*   Cbf       = (bf16_t*)ws;                     // 25,165,824 B
  size_t    off       = (size_t)BSZ * DIM * SEQB * sizeof(bf16_t);
  unsigned* rowMaxEnc = (unsigned*)(ws + off); off += (size_t)BSZ * SEQA * 4;
  unsigned* colMaxEnc = (unsigned*)(ws + off); off += (size_t)BSZ * SEQB * 4;
  float*    scoreA    = (float*)(ws + off);    off += (size_t)BSZ * SEQA * 4;
  float*    scoreB    = (float*)(ws + off);

  // rowMaxEnc and colMaxEnc are contiguous -> single init pass.
  init_kernel<<<(2 * BSZ * SEQA + 255) / 256, 256, 0, stream>>>(rowMaxEnc,
                                                                2 * BSZ * SEQA);
  dim3 g1(DIM / TM, SEQB / TN, BSZ);        // (6, 8, 16)
  gemm_ub_kernel<<<g1, 256, 0, stream>>>(U, Bin, Cbf);

  dim3 g2(SEQA / TM, SEQB / TN, BSZ);       // (8, 8, 16)
  gemm_align_kernel<<<g2, 256, 0, stream>>>(Ain, Cbf, Msk, rowMaxEnc, colMaxEnc);

  softmax_kernel<<<dim3(BSZ, 2), 256, 0, stream>>>(rowMaxEnc, colMaxEnc,
                                                   scoreA, scoreB);
  pool_kernel<<<dim3(DIM, BSZ, 2), 256, 0, stream>>>(Ain, Bin, scoreA, scoreB,
                                                     out);
}
